// Attention_6158983102987
// MI455X (gfx1250) — compile-verified, matrix operand order checked by reading, not run on hardware
//
#include <hip/hip_runtime.h>

// ---------------------------------------------------------------------------
// Types for CDNA5 WMMA (wave32): 16x16x32 bf16 -> f32
// ---------------------------------------------------------------------------
typedef __attribute__((ext_vector_type(16))) __bf16 v16bf;
typedef __attribute__((ext_vector_type(8)))  float  v8f;
typedef __attribute__((ext_vector_type(4)))  unsigned int ui32x4;
typedef __attribute__((ext_vector_type(8)))  int i32x8;
typedef __attribute__((ext_vector_type(4)))  int i32x4;

union BF16Frag {
  v16bf v;
  uint4 q[2];   // two 16-byte chunks: elements 0..7 and 8..15
};

__device__ __forceinline__ unsigned short f2bf(float f) {
  unsigned u = __float_as_uint(f);
  unsigned r = (u + 0x7FFFu + ((u >> 16) & 1u)) >> 16;   // round-to-nearest-even
  return (unsigned short)r;
}
__device__ __forceinline__ float bf2f(unsigned short s) {
  return __uint_as_float(((unsigned)s) << 16);
}

// Shapes (fixed by the reference)
#define BATCH   2
#define SEQ     2048
#define DMODEL  1024
#define NHEAD   16
#define DHEAD   64

// TDM availability (this toolchain: clang-23 6-arg form, probe-confirmed by
// the round-3 diagnostic "expected 6, have 5")
#if defined(__has_builtin)
#if __has_builtin(__builtin_amdgcn_tensor_load_to_lds)
#define HAVE_TDM 1
#endif
#endif
#ifndef HAVE_TDM
#define HAVE_TDM 0
#endif

// ---------------------------------------------------------------------------
// TDM: one instruction DMAs a 2-D bf16 tile (tile_d0 x tile_d1, row stride
// stride0 elements) from global memory into LDS.  D# packing per CDNA5 ISA
// CDNA5 ISA (group0: count/lds/global/type=image;
// group1: data_size=2B, tensor dims, tile dims, dim0 stride).
// ---------------------------------------------------------------------------
#if HAVE_TDM
__device__ __forceinline__ void tdm_load_2d(unsigned lds_byte,
                                            const unsigned short* gaddr,
                                            unsigned tensor_d0,  // row len (elems)
                                            unsigned tensor_d1,  // #rows in tensor
                                            unsigned tile_d0,    // tile width (elems)
                                            unsigned tile_d1,    // tile rows
                                            unsigned stride0) {  // row stride (elems)
  unsigned long long ga = (unsigned long long)(uintptr_t)gaddr;
  ui32x4 g0;
  g0[0] = 1u;                                        // count=1, user mode
  g0[1] = lds_byte;                                  // lds_addr (bytes)
  g0[2] = (unsigned)(ga & 0xFFFFFFFFu);              // global_addr[31:0]
  g0[3] = (unsigned)((ga >> 32) & 0x1FFFFFFu) | (2u << 30);  // [56:32] | type=2
  i32x8 g1;
  g1[0] = (int)(1u << 16);                           // data_size=1 (2 bytes)
  g1[1] = (int)((tensor_d0 & 0xFFFFu) << 16);        // tensor_dim0[15:0] @ bit48
  g1[2] = (int)((tensor_d0 >> 16) | ((tensor_d1 & 0xFFFFu) << 16));
  g1[3] = (int)((tensor_d1 >> 16) | (tile_d0 << 16));
  g1[4] = (int)tile_d1;                              // tile_dim1; tile_dim2=0
  g1[5] = (int)stride0;                              // tensor_dim0_stride[31:0]
  g1[6] = 0;
  g1[7] = 0;
  i32x4 gz4 = {0, 0, 0, 0};                          // 2-D: groups 2/3 unused
  i32x8 gz8 = {0, 0, 0, 0, 0, 0, 0, 0};
  __builtin_amdgcn_tensor_load_to_lds(g0, g1, gz4, gz4, gz8, 0);
}
#endif

// ---------------------------------------------------------------------------
// Elementwise f32 -> bf16
// ---------------------------------------------------------------------------
__global__ void k_f32_to_bf16(const float* __restrict__ in,
                              unsigned short* __restrict__ out, int n) {
  int i = blockIdx.x * blockDim.x + threadIdx.x;
  if (i < n) out[i] = f2bf(in[i]);
}

// Transpose K x N f32  ->  N x K bf16 (so GEMM B-fragments read contiguous K)
__global__ void k_transpose_to_bf16(const float* __restrict__ in,
                                    unsigned short* __restrict__ out,
                                    int K, int N) {
  int i = blockIdx.x * blockDim.x + threadIdx.x;
  if (i >= K * N) return;
  int k = i / N, n = i % N;
  out[(size_t)n * K + k] = f2bf(in[i]);
}

// ---------------------------------------------------------------------------
// Blocked WMMA GEMM, LDS double buffered.  Fill path: TDM tensor_load_to_lds
// (one DMA per 128x32 tile, TENSORcnt) when available, else per-lane
// GLOBAL_LOAD_ASYNC_TO_LDS_B128 (ASYNCcnt).
//   C(MxN) = A(MxK) * B(KxN), B supplied transposed (NxK, contiguous K).
// Workgroup (8 waves) computes a 128x128 macro-tile; each wave owns 64x32 =
// 4x2 accumulators, 8 v_wmma per 32-deep K-chunk.
// ---------------------------------------------------------------------------
#define GEMM_BM 128
#define GEMM_BN 128
#define GEMM_BK 32
#define ABUF_ELEMS (GEMM_BM * GEMM_BK)             // 4096 elems = 8KB
#define BUF_STRIDE (2 * ABUF_ELEMS)                // A + B per buffer

extern __shared__ unsigned short gemm_smem[];

__device__ __forceinline__ void async_fill_tile(const unsigned short* g_base,
                                                int ld, int row0, int k0,
                                                unsigned lds_byte_base, int t) {
#pragma unroll
  for (int p = 0; p < 2; ++p) {
    const int idx  = (p * 256 + t) * 16;   // byte index within the 8KB tile
    const int row  = idx >> 6;             // 64 bytes per row (32 bf16)
    const int colb = idx & 63;
    const char* ga =
        (const char*)(g_base + (size_t)(row0 + row) * ld + k0) + colb;
    const unsigned lo = lds_byte_base + (unsigned)idx;
    asm volatile("global_load_async_to_lds_b128 %0, %1, off"
                 :: "v"(lo), "v"((unsigned long long)(uintptr_t)ga)
                 : "memory");
  }
}

__global__ __launch_bounds__(256) void
k_gemm_bf16_wmma(const unsigned short* __restrict__ A,
                 const unsigned short* __restrict__ Bt,
                 unsigned short* __restrict__ outB,
                 float* __restrict__ outF,
                 const float* __restrict__ bias,
                 int M, int N, int K) {
  const int t    = threadIdx.x;
  const int lane = t & 31;
  const int wid  = t >> 5;
  const int half = lane >> 4;
  const int l16  = lane & 15;

  const int nbn = N / GEMM_BN;
  const int m0  = (blockIdx.x / nbn) * GEMM_BM;
  const int n0  = (blockIdx.x % nbn) * GEMM_BN;

  const int wr = wid >> 2;        // 0..1  -> M offset wr*64
  const int wc = wid & 3;         // 0..3  -> N offset wc*32

  // flat LDS aperture address low 32 bits == LDS byte offset (ISA 10.2)
  const unsigned smem_base =
      (unsigned)(unsigned long long)(uintptr_t)(void*)gemm_smem;

  v8f acc[4][2];
#pragma unroll
  for (int i = 0; i < 4; ++i)
#pragma unroll
    for (int j = 0; j < 2; ++j) acc[i][j] = {};

  const int nk = K / GEMM_BK;

  // prologue: fill buffer 0 with k-chunk 0
#if HAVE_TDM
  if (wid == 0) {
    tdm_load_2d(smem_base, A + (size_t)m0 * K, (unsigned)K, (unsigned)M,
                GEMM_BK, GEMM_BM, (unsigned)K);
    tdm_load_2d(smem_base + ABUF_ELEMS * 2, Bt + (size_t)n0 * K, (unsigned)K,
                (unsigned)N, GEMM_BK, GEMM_BM, (unsigned)K);
  }
#else
  async_fill_tile(A,  K, m0, 0, smem_base,                  t);
  async_fill_tile(Bt, K, n0, 0, smem_base + ABUF_ELEMS * 2, t);
#endif

  int cur = 0;
  for (int kc = 0; kc < nk; ++kc) {
    // wait for the in-flight DMA, then publish workgroup-wide
#if HAVE_TDM
    if (wid == 0) __builtin_amdgcn_s_wait_tensorcnt(0);
#else
    asm volatile("s_wait_asynccnt 0x0" ::: "memory");
#endif
    __syncthreads();

    // overlap: kick off next chunk's DMA into the other buffer
    if (kc + 1 < nk) {
      const int kn = (kc + 1) * GEMM_BK;
      const unsigned ob = smem_base + (unsigned)((1 - cur) * BUF_STRIDE) * 2;
#if HAVE_TDM
      if (wid == 0) {
        tdm_load_2d(ob, A + (size_t)m0 * K + kn, (unsigned)K, (unsigned)M,
                    GEMM_BK, GEMM_BM, (unsigned)K);
        tdm_load_2d(ob + ABUF_ELEMS * 2, Bt + (size_t)n0 * K + kn, (unsigned)K,
                    (unsigned)N, GEMM_BK, GEMM_BM, (unsigned)K);
      }
#else
      async_fill_tile(A,  K, m0, kn, ob,                  t);
      async_fill_tile(Bt, K, n0, kn, ob + ABUF_ELEMS * 2, t);
#endif
    }

    // fragment loads from LDS (ds_load_b128) + 8 WMMAs
    const unsigned abase = cur * BUF_STRIDE;
    const unsigned bbase = abase + ABUF_ELEMS;
    BF16Frag af[4], bf[2];
#pragma unroll
    for (int i = 0; i < 4; ++i) {
      const unsigned short* p =
          gemm_smem + abase + (unsigned)(wr * 64 + i * 16 + l16) * GEMM_BK +
          half * 8;
      af[i].q[0] = *(const uint4*)(p);
      af[i].q[1] = *(const uint4*)(p + 16);
    }
#pragma unroll
    for (int j = 0; j < 2; ++j) {
      const unsigned short* p =
          gemm_smem + bbase + (unsigned)(wc * 32 + j * 16 + l16) * GEMM_BK +
          half * 8;
      bf[j].q[0] = *(const uint4*)(p);
      bf[j].q[1] = *(const uint4*)(p + 16);
    }
#pragma unroll
    for (int i = 0; i < 4; ++i)
#pragma unroll
      for (int j = 0; j < 2; ++j)
        acc[i][j] = __builtin_amdgcn_wmma_f32_16x16x32_bf16(
            false, af[i].v, false, bf[j].v, (short)0, acc[i][j], false, false);

    cur ^= 1;
  }

  // epilogue
#pragma unroll
  for (int i = 0; i < 4; ++i) {
#pragma unroll
    for (int j = 0; j < 2; ++j) {
      const int n  = n0 + wc * 32 + j * 16 + l16;
      const float bv = bias ? bias[n] : 0.0f;
#pragma unroll
      for (int r = 0; r < 8; ++r) {
        const int m = m0 + wr * 64 + i * 16 + r + half * 8;
        if (outF) outF[(size_t)m * N + n] = acc[i][j][r] + bv;
        if (outB) outB[(size_t)m * N + n] = f2bf(acc[i][j][r]);
      }
    }
  }
}

// ---------------------------------------------------------------------------
// Per-head L2 normalization of Q,K + layout change; V stored d-major.
//   Qh/Kh : [b,h,n,d]   Vt : [b,h,d,n]
// ---------------------------------------------------------------------------
__global__ void k_norm_reshape(const unsigned short* __restrict__ Qb,
                               const unsigned short* __restrict__ Kb,
                               const unsigned short* __restrict__ Vb,
                               unsigned short* __restrict__ Qh,
                               unsigned short* __restrict__ Kh,
                               unsigned short* __restrict__ Vt) {
  const int lane = threadIdx.x & 31;
  const int wid  = threadIdx.x >> 5;
  const int g    = blockIdx.x * (blockDim.x >> 5) + wid;   // b*N*H groups
  const int h = g & (NHEAD - 1);
  const int n = (g >> 4) & (SEQ - 1);
  const int b = g >> 15;

  const size_t src = ((size_t)(b * SEQ + n)) * DMODEL + (size_t)h * DHEAD;
  float q0 = bf2f(Qb[src + lane]),      q1 = bf2f(Qb[src + lane + 32]);
  float k0 = bf2f(Kb[src + lane]),      k1 = bf2f(Kb[src + lane + 32]);
  float v0 = bf2f(Vb[src + lane]),      v1 = bf2f(Vb[src + lane + 32]);

  float sq = q0 * q0 + q1 * q1;
  float sk = k0 * k0 + k1 * k1;
#pragma unroll
  for (int m = 16; m >= 1; m >>= 1) {
    sq += __shfl_xor(sq, m, 32);
    sk += __shfl_xor(sk, m, 32);
  }
  const float rq = 1.0f / (sqrtf(sq) + 1e-6f);
  const float rk = 1.0f / (sqrtf(sk) + 1e-6f);

  const size_t bh    = (size_t)(b * NHEAD + h);
  const size_t dstqk = (bh * SEQ + n) * DHEAD;
  Qh[dstqk + lane]      = f2bf(q0 * rq);
  Qh[dstqk + lane + 32] = f2bf(q1 * rq);
  Kh[dstqk + lane]      = f2bf(k0 * rk);
  Kh[dstqk + lane + 32] = f2bf(k1 * rk);

  const size_t dstv = bh * DHEAD * SEQ;
  Vt[dstv + (size_t)lane * SEQ + n]        = f2bf(v0);
  Vt[dstv + (size_t)(lane + 32) * SEQ + n] = f2bf(v1);
}

// ---------------------------------------------------------------------------
// Flash attention (causal, online softmax), cooperative K/V staging.
// Block = 4 waves = 64 consecutive queries of one (b,h); wave w owns queries
// [qb0 + 16w, qb0 + 16w + 16).  Per 32-key tile the block async-DMAs the
// shared K tile (32x64) and V tile (64x32, d-major) into LDS (double
// buffered, ASYNCcnt handoff), then each wave runs 4 QK^T WMMAs + softmax +
// 4 PV WMMAs.  Tiles past a wave's causal range are fully masked (P=0), so
// the loop trip count is uniform and barriers are safe.
// ---------------------------------------------------------------------------
#define FLASH_WAVES 4

__device__ __forceinline__ void flash_fill(const unsigned short* Kbase,
                                           const unsigned short* Vbase,
                                           int k0, unsigned kb, unsigned vb,
                                           int t) {
#pragma unroll
  for (int p = 0; p < 2; ++p) {
    const int idx = (p * 128 + t) * 16;
    {  // K tile: 32 key-rows x 128B (64 bf16)
      const int row = idx >> 7, colb = idx & 127;
      const char* ga =
          (const char*)(Kbase + (size_t)(k0 + row) * DHEAD) + colb;
      const unsigned lo = kb + (unsigned)idx;
      asm volatile("global_load_async_to_lds_b128 %0, %1, off"
                   :: "v"(lo), "v"((unsigned long long)(uintptr_t)ga)
                   : "memory");
    }
    {  // V tile: 64 d-rows x 64B (32 bf16)
      const int row = idx >> 6, colb = idx & 63;
      const char* ga = (const char*)(Vbase + (size_t)row * SEQ + k0) + colb;
      const unsigned lo = vb + (unsigned)idx;
      asm volatile("global_load_async_to_lds_b128 %0, %1, off"
                   :: "v"(lo), "v"((unsigned long long)(uintptr_t)ga)
                   : "memory");
    }
  }
}

__global__ __launch_bounds__(FLASH_WAVES * 32) void
k_flash_attn(const unsigned short* __restrict__ Qh,
             const unsigned short* __restrict__ Kh,
             const unsigned short* __restrict__ Vt,
             unsigned short* __restrict__ Oh) {
  __shared__ unsigned short ksh[2][32 * DHEAD];          // 2 x 4KB
  __shared__ unsigned short vsh[2][DHEAD * 32];          // 2 x 4KB
  __shared__ unsigned short pbuf[FLASH_WAVES][16 * 32];  // 4KB

  const int t    = threadIdx.x;
  const int lane = t & 31;
  const int wid  = t >> 5;

  // block -> (b, h, 64-query strip)
  const int qb = blockIdx.x & 31;                 // SEQ/64 = 32
  const int h  = (blockIdx.x >> 5) & (NHEAD - 1);
  const int b  = blockIdx.x >> 9;
  const int q0 = qb * 64 + wid * 16;

  const int half = lane >> 4;
  const int l16  = lane & 15;
  const size_t bh = (size_t)(b * NHEAD + h);

  const unsigned short* Qbase = Qh + bh * SEQ * DHEAD;
  const unsigned short* Kbase = Kh + bh * SEQ * DHEAD;
  const unsigned short* Vbase = Vt + bh * DHEAD * SEQ;

  const unsigned kb0 = (unsigned)(unsigned long long)(uintptr_t)&ksh[0][0];
  const unsigned kb1 = (unsigned)(unsigned long long)(uintptr_t)&ksh[1][0];
  const unsigned vb0 = (unsigned)(unsigned long long)(uintptr_t)&vsh[0][0];
  const unsigned vb1 = (unsigned)(unsigned long long)(uintptr_t)&vsh[1][0];

  // Q fragments: 16x64 -> two 16x32 A-fragments (contiguous along d)
  BF16Frag qf[2];
  {
    const unsigned short* qrow = Qbase + (size_t)(q0 + l16) * DHEAD + half * 8;
    qf[0].q[0] = *(const uint4*)(qrow);
    qf[0].q[1] = *(const uint4*)(qrow + 16);
    qf[1].q[0] = *(const uint4*)(qrow + 32);
    qf[1].q[1] = *(const uint4*)(qrow + 48);
  }

  v8f o0 = {}, o1 = {}, o2 = {}, o3 = {};
  float mrow[8], lrow[8];
#pragma unroll
  for (int r = 0; r < 8; ++r) { mrow[r] = -1e30f; lrow[r] = 0.0f; }

  const float SCALE = 0.125f;         // 1/sqrt(DHEAD)
  const float LOG2E = 1.44269504f;

  const int nk = (qb * 64 + 64) / 32; // uniform across the block

  flash_fill(Kbase, Vbase, 0, kb0, vb0, t);

  int cur = 0;
  for (int kc = 0; kc < nk; ++kc) {
    const int k0 = kc * 32;

    asm volatile("s_wait_asynccnt 0x0" ::: "memory");
    __syncthreads();

    if (kc + 1 < nk)
      flash_fill(Kbase, Vbase, k0 + 32, cur ? kb0 : kb1, cur ? vb0 : vb1, t);

    // ---- S = Q * K^T : two 16x16 accumulators from shared K tile ----
    v8f s0 = {}, s1 = {};
#pragma unroll
    for (int kcc = 0; kcc < 2; ++kcc) {
      BF16Frag kf0, kf1;
      const unsigned short* kr0 =
          &ksh[cur][(unsigned)l16 * DHEAD + kcc * 32 + half * 8];
      const unsigned short* kr1 = kr0 + 16 * DHEAD;
      kf0.q[0] = *(const uint4*)(kr0);
      kf0.q[1] = *(const uint4*)(kr0 + 16);
      kf1.q[0] = *(const uint4*)(kr1);
      kf1.q[1] = *(const uint4*)(kr1 + 16);
      s0 = __builtin_amdgcn_wmma_f32_16x16x32_bf16(false, qf[kcc].v, false,
                                                   kf0.v, (short)0, s0, false,
                                                   false);
      s1 = __builtin_amdgcn_wmma_f32_16x16x32_bf16(false, qf[kcc].v, false,
                                                   kf1.v, (short)0, s1, false,
                                                   false);
    }

    // ---- causal mask + scale + online softmax update ----
    const bool need_mask = (k0 + 31 > q0);
#pragma unroll
    for (int r = 0; r < 8; ++r) {
      const int qrow = q0 + r + half * 8;
      float a0 = s0[r] * SCALE;
      float a1 = s1[r] * SCALE;
      if (need_mask) {
        if (k0 + l16 > qrow)      a0 = -1e30f;
        if (k0 + 16 + l16 > qrow) a1 = -1e30f;
      }
      float mx = fmaxf(a0, a1);
      mx = fmaxf(mx, __shfl_xor(mx, 1, 32));
      mx = fmaxf(mx, __shfl_xor(mx, 2, 32));
      mx = fmaxf(mx, __shfl_xor(mx, 4, 32));
      mx = fmaxf(mx, __shfl_xor(mx, 8, 32));
      const float mn    = fmaxf(mrow[r], mx);
      const float alpha = exp2f((mrow[r] - mn) * LOG2E);
      const float p0 = exp2f((a0 - mn) * LOG2E);
      const float p1 = exp2f((a1 - mn) * LOG2E);
      float rs = p0 + p1;
      rs += __shfl_xor(rs, 1, 32);
      rs += __shfl_xor(rs, 2, 32);
      rs += __shfl_xor(rs, 4, 32);
      rs += __shfl_xor(rs, 8, 32);
      lrow[r] = lrow[r] * alpha + rs;
      mrow[r] = mn;

      const int row = r + half * 8;
      pbuf[wid][row * 32 + l16]      = f2bf(p0);
      pbuf[wid][row * 32 + 16 + l16] = f2bf(p1);

      o0[r] *= alpha; o1[r] *= alpha; o2[r] *= alpha; o3[r] *= alpha;
    }

    // same-wave LDS store -> load ordering (DS ops are in-order per wave)
    __builtin_amdgcn_sched_barrier(0);
    asm volatile("s_wait_dscnt 0x0" ::: "memory");

    BF16Frag pf;
    {
      const unsigned short* prow = &pbuf[wid][l16 * 32 + half * 8];
      pf.q[0] = *(const uint4*)(prow);
      pf.q[1] = *(const uint4*)(prow + 16);
    }

    // ---- O += P * V from shared V tile (d-major, contiguous along keys) ----
#define PV_STEP(DT, OACC)                                                     \
    do {                                                                      \
      BF16Frag vf;                                                            \
      const unsigned short* vrow =                                            \
          &vsh[cur][(unsigned)((DT) * 16 + l16) * 32 + half * 8];             \
      vf.q[0] = *(const uint4*)(vrow);                                        \
      vf.q[1] = *(const uint4*)(vrow + 16);                                   \
      OACC = __builtin_amdgcn_wmma_f32_16x16x32_bf16(false, pf.v, false, vf.v,\
                                                     (short)0, OACC, false,   \
                                                     false);                  \
    } while (0)
    PV_STEP(0, o0);
    PV_STEP(1, o1);
    PV_STEP(2, o2);
    PV_STEP(3, o3);
#undef PV_STEP

    cur ^= 1;
  }

  // ---- epilogue: divide by softmax denominator, store to [b,n,h*64+d] ----
  unsigned short* obase =
      Oh + ((size_t)(b * SEQ + q0)) * DMODEL + (size_t)h * DHEAD;
#pragma unroll
  for (int r = 0; r < 8; ++r) {
    const int row   = r + half * 8;
    const float inv = 1.0f / lrow[r];
    obase[(size_t)row * DMODEL +  0 + l16] = f2bf(o0[r] * inv);
    obase[(size_t)row * DMODEL + 16 + l16] = f2bf(o1[r] * inv);
    obase[(size_t)row * DMODEL + 32 + l16] = f2bf(o2[r] * inv);
    obase[(size_t)row * DMODEL + 48 + l16] = f2bf(o3[r] * inv);
  }
}

// ---------------------------------------------------------------------------
// Host-side orchestration
// ---------------------------------------------------------------------------
extern "C" void kernel_launch(void* const* d_in, const int* in_sizes, int n_in,
                              void* d_out, int out_size, void* d_ws,
                              size_t ws_size, hipStream_t stream) {
  const float* X  = (const float*)d_in[0];
  const float* Wq = (const float*)d_in[1];
  const float* Wk = (const float*)d_in[2];
  const float* Wv = (const float*)d_in[3];
  const float* Wo = (const float*)d_in[4];
  const float* bo = (const float*)d_in[5];
  float* out = (float*)d_out;

  const int BN = BATCH * SEQ;              // 4096 rows

  char* ws = (char*)d_ws;
  size_t off = 0;
  auto alloc = [&](size_t bytes) -> char* {
    char* p = ws + off;
    off += (bytes + 255) & ~(size_t)255;
    return p;
  };
  unsigned short* Xb  = (unsigned short*)alloc((size_t)BN * DMODEL * 2);
  unsigned short* WqT = (unsigned short*)alloc((size_t)DMODEL * DMODEL * 2);
  unsigned short* WkT = (unsigned short*)alloc((size_t)DMODEL * DMODEL * 2);
  unsigned short* WvT = (unsigned short*)alloc((size_t)DMODEL * DMODEL * 2);
  unsigned short* WoT = (unsigned short*)alloc((size_t)DMODEL * DMODEL * 2);
  unsigned short* Qb  = (unsigned short*)alloc((size_t)BN * DMODEL * 2);
  unsigned short* Kb  = (unsigned short*)alloc((size_t)BN * DMODEL * 2);
  unsigned short* Vb  = (unsigned short*)alloc((size_t)BN * DMODEL * 2);
  unsigned short* Qh  = (unsigned short*)alloc((size_t)BN * DMODEL * 2);
  unsigned short* Khd = (unsigned short*)alloc((size_t)BN * DMODEL * 2);
  unsigned short* Vt  = (unsigned short*)alloc((size_t)BN * DMODEL * 2);
  unsigned short* Oh  = (unsigned short*)alloc((size_t)BN * DMODEL * 2);
  (void)ws_size; (void)in_sizes; (void)n_in; (void)out_size;

  // 1) convert X to bf16; transpose-convert all weights
  {
    const int n = BN * DMODEL;
    k_f32_to_bf16<<<n / 256, 256, 0, stream>>>(X, Xb, n);
    const int wn = DMODEL * DMODEL;
    k_transpose_to_bf16<<<wn / 256, 256, 0, stream>>>(Wq, WqT, DMODEL, DMODEL);
    k_transpose_to_bf16<<<wn / 256, 256, 0, stream>>>(Wk, WkT, DMODEL, DMODEL);
    k_transpose_to_bf16<<<wn / 256, 256, 0, stream>>>(Wv, WvT, DMODEL, DMODEL);
    k_transpose_to_bf16<<<wn / 256, 256, 0, stream>>>(Wo, WoT, DMODEL, DMODEL);
  }

  // 2) QKV projections: 4096x1024x1024 bf16 WMMA GEMMs (TDM/async staged)
  {
    const int blocks = (BN / GEMM_BM) * (DMODEL / GEMM_BN);     // 256
    const size_t lds = 2 * BUF_STRIDE * sizeof(unsigned short); // 32KB
    k_gemm_bf16_wmma<<<blocks, 256, lds, stream>>>(Xb, WqT, Qb, nullptr,
                                                   nullptr, BN, DMODEL, DMODEL);
    k_gemm_bf16_wmma<<<blocks, 256, lds, stream>>>(Xb, WkT, Kb, nullptr,
                                                   nullptr, BN, DMODEL, DMODEL);
    k_gemm_bf16_wmma<<<blocks, 256, lds, stream>>>(Xb, WvT, Vb, nullptr,
                                                   nullptr, BN, DMODEL, DMODEL);
  }

  // 3) per-head L2 norm of Q,K + reshape; V to d-major layout
  {
    const int groups = BATCH * SEQ * NHEAD;         // 65536 waves
    k_norm_reshape<<<groups / 8, 256, 0, stream>>>(Qb, Kb, Vb, Qh, Khd, Vt);
  }

  // 4) causal flash attention (cooperative async K/V staging + WMMA)
  {
    const int blocks = BATCH * NHEAD * (SEQ / 64);  // 1024
    k_flash_attn<<<blocks, FLASH_WAVES * 32, 0, stream>>>(Qh, Khd, Vt, Oh);
  }

  // 5) output projection + bias -> f32
  {
    const int blocks = (BN / GEMM_BM) * (DMODEL / GEMM_BN);
    const size_t lds = 2 * BUF_STRIDE * sizeof(unsigned short);
    k_gemm_bf16_wmma<<<blocks, 256, lds, stream>>>(Oh, WoT, nullptr, out, bo,
                                                   BN, DMODEL, DMODEL);
  }
}
